// MHA_17721035064131
// MI455X (gfx1250) — matrix operand order repaired and checked by reading, not compiled, not run on hardware
//
#include <hip/hip_runtime.h>

#define HIDDEN 768
#define HEADS  12
#define HDIM   64
#define SEQ    512
#define DSK    64
#define NH     768   // HEADS*HDIM
#define BATCH  32
#define EPSF   1e-6f

typedef __bf16 bf16;
typedef __attribute__((ext_vector_type(16))) __bf16 v16bf;
typedef __attribute__((ext_vector_type(8)))  __bf16 v8bf;
typedef __attribute__((ext_vector_type(4)))  __bf16 v4bf;
typedef __attribute__((ext_vector_type(8)))  float  v8f;
typedef __attribute__((ext_vector_type(4)))  float  v4f;

union ABf { v16bf v; v8bf h[2]; };

// A-tile: 16(M) x 32(K) bf16 from row-major [M][K] (ld in elements).
// lane<16: M=lane, K halves {0..7,16..23}; lane>=16: M=lane-16, K {8..15,24..31}
__device__ __forceinline__ v16bf load_tile_A(const bf16* p, int ld, int lane) {
  const int half = lane >> 4, row = lane & 15;
  ABf t;
  t.h[0] = *(const v8bf*)(p + (size_t)row * ld + half * 8);
  t.h[1] = *(const v8bf*)(p + (size_t)row * ld + half * 8 + 16);
  return t.v;
}
// B-tile: 32(K) x 16(N) bf16 from [N][K] storage (pre-transposed, ld in elements).
// lane<16: N=lane, K {0..7,16..23}; lane>=16: N=lane-16, K {8..15,24..31}
__device__ __forceinline__ v16bf load_tile_B(const bf16* p, int ld, int lane) {
  const int half = lane >> 4, row = lane & 15;
  ABf t;
  t.h[0] = *(const v8bf*)(p + (size_t)row * ld + half * 8);
  t.h[1] = *(const v8bf*)(p + (size_t)row * ld + half * 8 + 16);
  return t.v;
}
#define WMMA_BF16(a, b, c) \
  __builtin_amdgcn_wmma_f32_16x16x32_bf16(false, (a), false, (b), (short)0, (c), false, false)

__device__ __forceinline__ float elu1(float x) { return x > 0.0f ? x + 1.0f : __expf(x); }

// ---------------- weight prep (small, one-shot) ----------------
__global__ void k_w_qkv(const float* __restrict__ Wq, const float* __restrict__ Wk,
                        const float* __restrict__ Wv, bf16* __restrict__ Wt) {
  int idx = blockIdx.x * blockDim.x + threadIdx.x;
  if (idx >= 3 * HIDDEN * NH) return;
  int i = idx / (HIDDEN * NH), rem = idx % (HIDDEN * NH);
  int d = rem / NH, nh = rem % NH;
  const float* W = (i == 0) ? Wq : (i == 1) ? Wk : Wv;
  Wt[(size_t)i * NH * HIDDEN + (size_t)nh * HIDDEN + d] = (bf16)W[rem];
}
__global__ void k_w_out(const float* __restrict__ Wout, bf16* __restrict__ WoutT) {
  int idx = blockIdx.x * blockDim.x + threadIdx.x;
  if (idx >= NH * HIDDEN) return;
  int nh = idx / HIDDEN, D = idx % HIDDEN;
  WoutT[(size_t)D * NH + nh] = (bf16)Wout[idx];
}
__global__ void k_w_ds(const float* __restrict__ Wk, const float* __restrict__ Wv,
                       bf16* __restrict__ Wds) {
  int idx = blockIdx.x * blockDim.x + threadIdx.x;
  if (idx >= 2 * DSK * SEQ) return;
  int which = idx / (DSK * SEQ), rem = idx % (DSK * SEQ);
  Wds[idx] = (bf16)((which ? Wv : Wk)[rem]);
}
// ---------------- input conversion ----------------
__global__ void k_cvt_bf16_vec4(const float* __restrict__ in, bf16* __restrict__ out, int n4) {
  int i = blockIdx.x * blockDim.x + threadIdx.x;
  if (i >= n4) return;
  v4f f = *(const v4f*)(in + (size_t)i * 4);
  v4bf o;
  o.x = (bf16)f.x; o.y = (bf16)f.y; o.z = (bf16)f.z; o.w = (bf16)f.w;
  *(v4bf*)(out + (size_t)i * 4) = o;
}
// key/value f32 [b][s][d] -> bf16 [b][d][s] (tiled LDS transpose)
__global__ void k_transpose_kv(const float* __restrict__ key, const float* __restrict__ val,
                               bf16* __restrict__ kT, bf16* __restrict__ vT) {
  __shared__ float tile[32][33];
  const int dt = blockIdx.x * 32, st = blockIdx.y * 32;
  const int b = blockIdx.z >> 1, which = blockIdx.z & 1;
  const float* src = (which ? val : key) + (size_t)b * SEQ * HIDDEN;
  bf16* dst = (which ? vT : kT) + (size_t)b * HIDDEN * SEQ;
  const int tx = threadIdx.x & 31, ty = threadIdx.x >> 5;
  for (int i = 0; i < 4; ++i)
    tile[ty + i * 8][tx] = src[(size_t)(st + ty + i * 8) * HIDDEN + dt + tx];
  __syncthreads();
  for (int i = 0; i < 4; ++i)
    dst[(size_t)(dt + ty + i * 8) * SEQ + st + tx] = (bf16)tile[tx][ty + i * 8];
}

// ---------------- GEMM 1: downsample  kds[b][k][d] = Wds @ K/V ----------------
// grid (12, B, 2), block 128 (4 waves). A=[64][512] Wds, B=[d][s] kT/vT.
__global__ void k_gemm_ds(const bf16* __restrict__ Wds, const bf16* __restrict__ kT,
                          const bf16* __restrict__ vTin, bf16* __restrict__ kds,
                          bf16* __restrict__ vds) {
  const int lane = threadIdx.x & 31, wave = threadIdx.x >> 5;
  const int b = blockIdx.y, which = blockIdx.z;
  const int ncol0 = blockIdx.x * 64 + wave * 16;  // d
  const bf16* A  = Wds + (size_t)which * DSK * SEQ;
  const bf16* Bm = (which ? vTin : kT) + (size_t)b * HIDDEN * SEQ;
  v8f c[4] = {};
  for (int kk = 0; kk < SEQ / 32; ++kk) {
    v16bf bm = load_tile_B(Bm + (size_t)ncol0 * SEQ + kk * 32, SEQ, lane);
#pragma unroll
    for (int mt = 0; mt < 4; ++mt) {
      v16bf am = load_tile_A(A + (size_t)(mt * 16) * SEQ + kk * 32, SEQ, lane);
      c[mt] = WMMA_BF16(am, bm, c[mt]);
    }
  }
  bf16* out = (which ? vds : kds) + (size_t)b * DSK * HIDDEN;
  const int half = lane >> 4, col = ncol0 + (lane & 15);
#pragma unroll
  for (int mt = 0; mt < 4; ++mt)
    for (int r = 0; r < 8; ++r)
      out[(size_t)(mt * 16 + half * 8 + r) * HIDDEN + col] = (bf16)c[mt][r];
}

// ---------------- GEMM 2: K/V projections, epilogue bias(+elu), store transposed ----------------
// grid (12, B, 2), block 128. Output phiKT/vT: [b][nh][k] bf16 (packed b128 stores).
__global__ void k_gemm_proj(const bf16* __restrict__ kds, const bf16* __restrict__ vds,
                            const bf16* __restrict__ Wt, const float* __restrict__ bk,
                            const float* __restrict__ bv, bf16* __restrict__ phiKT,
                            bf16* __restrict__ vT) {
  const int lane = threadIdx.x & 31, wave = threadIdx.x >> 5;
  const int b = blockIdx.y, which = blockIdx.z;
  const int ncol0 = blockIdx.x * 64 + wave * 16;  // nh
  const bf16* A  = (which ? vds : kds) + (size_t)b * DSK * HIDDEN;
  const bf16* Bm = Wt + (size_t)(which + 1) * NH * HIDDEN;  // Wk_t or Wv_t
  v8f c[4] = {};
  for (int kk = 0; kk < HIDDEN / 32; ++kk) {
    v16bf bm = load_tile_B(Bm + (size_t)ncol0 * HIDDEN + kk * 32, HIDDEN, lane);
#pragma unroll
    for (int mt = 0; mt < 4; ++mt) {
      v16bf am = load_tile_A(A + (size_t)(mt * 16) * HIDDEN + kk * 32, HIDDEN, lane);
      c[mt] = WMMA_BF16(am, bm, c[mt]);
    }
  }
  const int half = lane >> 4, nh = ncol0 + (lane & 15);
  const float bias = (which ? bv : bk)[nh];
  bf16* out = (which ? vT : phiKT) + (size_t)b * NH * DSK;
#pragma unroll
  for (int mt = 0; mt < 4; ++mt) {
    v8bf pk;
    for (int r = 0; r < 8; ++r) {
      float v = c[mt][r] + bias;
      if (!which) v = elu1(v);
      pk[r] = (bf16)v;
    }
    *(v8bf*)(out + (size_t)nh * DSK + mt * 16 + half * 8) = pk;
  }
}

// ---------------- GEMM 3: Q projection, epilogue elu+1, store [b][q][nh] ----------------
// grid (12, 8, B), block 128.
__global__ void k_gemm_q(const bf16* __restrict__ qbf, const bf16* __restrict__ Wt,
                         const float* __restrict__ bq, bf16* __restrict__ phiQ) {
  const int lane = threadIdx.x & 31, wave = threadIdx.x >> 5;
  const int ncol0 = blockIdx.x * 64 + wave * 16;  // nh
  const int qbase = blockIdx.y * 64, b = blockIdx.z;
  const bf16* A = qbf + (size_t)b * SEQ * HIDDEN;
  v8f c[4] = {};
  for (int kk = 0; kk < HIDDEN / 32; ++kk) {
    v16bf bm = load_tile_B(Wt + (size_t)ncol0 * HIDDEN + kk * 32, HIDDEN, lane);
#pragma unroll
    for (int mt = 0; mt < 4; ++mt) {
      v16bf am = load_tile_A(A + (size_t)(qbase + mt * 16) * HIDDEN + kk * 32, HIDDEN, lane);
      c[mt] = WMMA_BF16(am, bm, c[mt]);
    }
  }
  const int half = lane >> 4, nh = ncol0 + (lane & 15);
  const float bias = bq[nh];
#pragma unroll
  for (int mt = 0; mt < 4; ++mt)
    for (int r = 0; r < 8; ++r) {
      int q = qbase + mt * 16 + half * 8 + r;
      phiQ[((size_t)b * SEQ + q) * NH + nh] = (bf16)elu1(c[mt][r] + bias);
    }
}

// ---------------- GEMM 4: KV state s = phiK^T @ V per (b,n), plus z ----------------
// grid (12, B), block 64 (2 waves). Stores sT[b][n][h][d] bf16, z[b][n][d] f32.
__global__ void k_s_z(const bf16* __restrict__ phiKT, const bf16* __restrict__ vT,
                      bf16* __restrict__ sT, float* __restrict__ zbuf) {
  const int n = blockIdx.x, b = blockIdx.y;
  const int lane = threadIdx.x & 31, wave = threadIdx.x >> 5;
  const bf16* A  = phiKT + ((size_t)b * NH + n * HDIM) * DSK;  // [d][k]
  const bf16* Bm = vT    + ((size_t)b * NH + n * HDIM) * DSK;  // [h][k]
  v8f c[4][2] = {};
  for (int kk = 0; kk < 2; ++kk) {
    v16bf bm[2];
    for (int nt = 0; nt < 2; ++nt)
      bm[nt] = load_tile_B(Bm + (size_t)(wave * 32 + nt * 16) * DSK + kk * 32, DSK, lane);
#pragma unroll
    for (int mt = 0; mt < 4; ++mt) {
      v16bf am = load_tile_A(A + (size_t)(mt * 16) * DSK + kk * 32, DSK, lane);
      for (int nt = 0; nt < 2; ++nt) c[mt][nt] = WMMA_BF16(am, bm[nt], c[mt][nt]);
    }
  }
  const int half = lane >> 4;
  bf16* sTo = sT + (size_t)(b * HEADS + n) * HDIM * HDIM;
#pragma unroll
  for (int mt = 0; mt < 4; ++mt)
    for (int nt = 0; nt < 2; ++nt) {
      int h = wave * 32 + nt * 16 + (lane & 15);
      v8bf pk;
      for (int r = 0; r < 8; ++r) pk[r] = (bf16)c[mt][nt][r];
      *(v8bf*)(sTo + (size_t)h * HDIM + mt * 16 + half * 8) = pk;
    }
  // z[d] = sum_k phiK[k][d]  (row sums of [d][k])
  const bf16* row = A + (size_t)threadIdx.x * DSK;
  float sum = 0.f;
  for (int k = 0; k < DSK; ++k) sum += (float)row[k];
  zbuf[(size_t)(b * HEADS + n) * HDIM + threadIdx.x] = sum;
}

// ---------------- GEMM 5 (fused): qs, 1/qz normalize, out projection ----------------
// grid (8, B), block 192 (6 waves). LDS: a_v [64][768] bf16 + qz [12][64] f32.
__global__ __launch_bounds__(192) void k_final(const bf16* __restrict__ phiQ,
                                               const bf16* __restrict__ sT,
                                               const float* __restrict__ zbuf,
                                               const bf16* __restrict__ WoutT,
                                               const float* __restrict__ bout,
                                               float* __restrict__ out) {
  extern __shared__ char smem[];
  bf16*  av   = (bf16*)smem;                                 // [64][NH]
  float* qzsh = (float*)(smem + (size_t)64 * NH * sizeof(bf16));  // [12][64]
  const int lane = threadIdx.x & 31, wave = threadIdx.x >> 5;
  const int half = lane >> 4;
  const int qbase = blockIdx.x * 64, b = blockIdx.y;

  for (int hh = 0; hh < 2; ++hh) {
    const int n = wave + 6 * hh;
    // qz = eps + phiQ . z  (2 rows per lane)
    const float* zp = zbuf + (size_t)(b * HEADS + n) * HDIM;
    for (int j = 0; j < 2; ++j) {
      int qloc = lane * 2 + j;
      const bf16* pq = phiQ + ((size_t)(b * SEQ + qbase + qloc)) * NH + n * HDIM;
      float s = EPSF;
      for (int d = 0; d < HDIM; ++d) s += (float)pq[d] * zp[d];
      qzsh[n * HDIM + qloc] = s;
    }
    // qs = phiQ @ s : 64x64x64
    v8f c[4][4] = {};
    const bf16* sb = sT + (size_t)(b * HEADS + n) * HDIM * HDIM;  // [h][d]
    for (int kk = 0; kk < 2; ++kk) {
      v16bf bm[4];
      for (int nt = 0; nt < 4; ++nt)
        bm[nt] = load_tile_B(sb + (size_t)(nt * 16) * HDIM + kk * 32, HDIM, lane);
#pragma unroll
      for (int mt = 0; mt < 4; ++mt) {
        v16bf am = load_tile_A(
            phiQ + ((size_t)(b * SEQ + qbase + mt * 16)) * NH + n * HDIM + kk * 32, NH, lane);
        for (int nt = 0; nt < 4; ++nt) c[mt][nt] = WMMA_BF16(am, bm[nt], c[mt][nt]);
      }
    }
    // normalize and stash a_v into LDS as [q][nh]
    for (int mt = 0; mt < 4; ++mt)
      for (int nt = 0; nt < 4; ++nt) {
        int h = nt * 16 + (lane & 15);
        for (int r = 0; r < 8; ++r) {
          int qloc = mt * 16 + half * 8 + r;
          av[(size_t)qloc * NH + n * HDIM + h] = (bf16)(c[mt][nt][r] / qzsh[n * HDIM + qloc]);
        }
      }
  }
  __syncthreads();
  // out[q][D] = a_v @ Wout : 64 x 768 x 768; wave covers D in [128w,128w+128), 2 passes
  for (int pass = 0; pass < 2; ++pass) {
    const int Dbase = wave * 128 + pass * 64;
    v8f c[4][4] = {};
    for (int kk = 0; kk < NH / 32; ++kk) {
      v16bf bm[4];
      for (int nt = 0; nt < 4; ++nt)
        bm[nt] = load_tile_B(WoutT + (size_t)(Dbase + nt * 16) * NH + kk * 32, NH, lane);
#pragma unroll
      for (int mt = 0; mt < 4; ++mt) {
        v16bf am = load_tile_A(av + (size_t)(mt * 16) * NH + kk * 32, NH, lane);
        for (int nt = 0; nt < 4; ++nt) c[mt][nt] = WMMA_BF16(am, bm[nt], c[mt][nt]);
      }
    }
    for (int mt = 0; mt < 4; ++mt)
      for (int nt = 0; nt < 4; ++nt) {
        int D = Dbase + nt * 16 + (lane & 15);
        float bb = bout[D];
        for (int r = 0; r < 8; ++r) {
          int q = qbase + mt * 16 + half * 8 + r;
          out[((size_t)(b * SEQ + q)) * HIDDEN + D] = c[mt][nt][r] + bb;
        }
      }
  }
}

extern "C" void kernel_launch(void* const* d_in, const int* in_sizes, int n_in,
                              void* d_out, int out_size, void* d_ws, size_t ws_size,
                              hipStream_t stream) {
  const float* query = (const float*)d_in[0];
  const float* key   = (const float*)d_in[1];
  const float* value = (const float*)d_in[2];
  // d_in[3] = step (unused, mask=False path)
  const float* W_kds = (const float*)d_in[4];
  const float* W_vds = (const float*)d_in[5];
  const float* Wq    = (const float*)d_in[6];
  const float* bq    = (const float*)d_in[7];
  const float* Wk    = (const float*)d_in[8];
  const float* bk    = (const float*)d_in[9];
  const float* Wv    = (const float*)d_in[10];
  const float* bv    = (const float*)d_in[11];
  const float* Wout  = (const float*)d_in[12];
  const float* bout  = (const float*)d_in[13];
  float* out = (float*)d_out;

  size_t off = 0;
  char* ws = (char*)d_ws;
  auto carve = [&](size_t bytes) -> void* {
    void* p = ws + off;
    off += (bytes + 255) & ~(size_t)255;
    return p;
  };
  bf16* qbf   = (bf16*)carve((size_t)BATCH * SEQ * HIDDEN * 2);
  bf16* kT    = (bf16*)carve((size_t)BATCH * HIDDEN * SEQ * 2);
  bf16* vTin  = (bf16*)carve((size_t)BATCH * HIDDEN * SEQ * 2);
  bf16* Wt    = (bf16*)carve((size_t)3 * NH * HIDDEN * 2);
  bf16* WoutT = (bf16*)carve((size_t)HIDDEN * NH * 2);
  bf16* Wds   = (bf16*)carve((size_t)2 * DSK * SEQ * 2);
  bf16* kds   = (bf16*)carve((size_t)BATCH * DSK * HIDDEN * 2);
  bf16* vds   = (bf16*)carve((size_t)BATCH * DSK * HIDDEN * 2);
  bf16* phiKT = (bf16*)carve((size_t)BATCH * NH * DSK * 2);
  bf16* vTp   = (bf16*)carve((size_t)BATCH * NH * DSK * 2);
  bf16* sTb   = (bf16*)carve((size_t)BATCH * HEADS * HDIM * HDIM * 2);
  float* zb   = (float*)carve((size_t)BATCH * HEADS * HDIM * 4);
  bf16* phiQ  = (bf16*)carve((size_t)BATCH * SEQ * NH * 2);
  (void)ws_size; (void)in_sizes; (void)n_in; (void)out_size;

  // weight prep
  k_w_qkv<<<(3 * HIDDEN * NH + 255) / 256, 256, 0, stream>>>(Wq, Wk, Wv, Wt);
  k_w_out<<<(NH * HIDDEN + 255) / 256, 256, 0, stream>>>(Wout, WoutT);
  k_w_ds<<<(2 * DSK * SEQ + 255) / 256, 256, 0, stream>>>(W_kds, W_vds, Wds);
  // input conversion
  int n4 = BATCH * SEQ * HIDDEN / 4;
  k_cvt_bf16_vec4<<<(n4 + 255) / 256, 256, 0, stream>>>(query, qbf, n4);
  k_transpose_kv<<<dim3(HIDDEN / 32, SEQ / 32, BATCH * 2), 256, 0, stream>>>(key, value, kT, vTin);
  // pipeline
  k_gemm_ds<<<dim3(12, BATCH, 2), 128, 0, stream>>>(Wds, kT, vTin, kds, vds);
  k_gemm_proj<<<dim3(12, BATCH, 2), 128, 0, stream>>>(kds, vds, Wt, bk, bv, phiKT, vTp);
  k_gemm_q<<<dim3(12, SEQ / 64, BATCH), 128, 0, stream>>>(qbf, Wt, bq, phiQ);
  k_s_z<<<dim3(HEADS, BATCH), 64, 0, stream>>>(phiKT, vTp, sTb, zb);
  size_t smem = (size_t)64 * NH * sizeof(bf16) + (size_t)HEADS * HDIM * sizeof(float);
  k_final<<<dim3(SEQ / 64, BATCH), 192, smem, stream>>>(phiQ, sTb, zb, WoutT, bout, out);
}
